// ResidualGRULayer_50294067036292
// MI455X (gfx1250) — matrix operand order repaired and checked by reading, not compiled, not run on hardware
//
#include <hip/hip_runtime.h>
#include <hip/hip_bf16.h>
#include <math.h>

// ---------------------------------------------------------------------------
// Problem constants (match reference)
// ---------------------------------------------------------------------------
#define B_   64
#define T_   512            // power of two: time index = row & (T_-1)
#define F_   256
#define U_   512
#define M_   (B_ * T_)      // 32768 rows
#define ALPHA_ 0.2f
#define EPS_   1e-3f
#define NB_SCAN 16          // persistent blocks for GRU scan (128 waves)

typedef __attribute__((ext_vector_type(16))) __bf16 v16bf;
typedef __attribute__((ext_vector_type(8)))  __bf16 v8bf;
typedef __attribute__((ext_vector_type(8)))  float  v8f;

// ---------------------------------------------------------------------------
// Fragment helpers (CDNA5 wave32 WMMA layouts, 16-bit data)
// ---------------------------------------------------------------------------
__device__ __forceinline__ v16bf load_a_frag(const __bf16* p) {
  union { v16bf v; v8bf h[2]; } u;
  u.h[0] = *(const v8bf*)(p);
  u.h[1] = *(const v8bf*)(p + 8);
  return u.v;
}

__device__ __forceinline__ v8f wmma_bf16(v16bf a, v16bf b, v8f c) {
  return __builtin_amdgcn_wmma_f32_16x16x32_bf16(
      /*neg_a=*/false, a, /*neg_b=*/false, b,
      /*c_mod=*/(short)0, c, /*reuse_a=*/false, /*reuse_b=*/false);
}

// Fast activations (exp + hardware rcp; matches bf16-path precision class)
__device__ __forceinline__ float fast_sigmoid(float x) {
  return __builtin_amdgcn_rcpf(1.f + __expf(-x));
}
__device__ __forceinline__ float fast_tanh(float x) {
  float e = __expf(-2.f * x);
  return (1.f - e) * __builtin_amdgcn_rcpf(1.f + e);
}

// ---------------------------------------------------------------------------
// Small prep kernels
// ---------------------------------------------------------------------------
__global__ void f32_to_bf16_k(const float* __restrict__ x, __bf16* __restrict__ y, int n) {
  int i = blockIdx.x * 256 + threadIdx.x;
  if (i < n) y[i] = (__bf16)x[i];
}

__global__ void zero_bf16_k(__bf16* __restrict__ p, int n) {
  int i = blockIdx.x * 256 + threadIdx.x;
  if (i < n) p[i] = (__bf16)0.f;
}

// W row-major [K][N] (fp32) -> Bp [N][K] bf16 (B-transposed pack)
__global__ void pack_transpose_k(const float* __restrict__ W, __bf16* __restrict__ Bp,
                                 int K, int N) {
  int i = blockIdx.x * 256 + threadIdx.x;
  if (i >= K * N) return;
  int k = i / N, n = i - k * N;
  Bp[(size_t)n * K + k] = (__bf16)W[i];
}

// s = g*rsqrt(v+eps); t = (bias - m)*s + beta   (conv bias folded into BN)
__global__ void prep_bn_k(const float* __restrict__ g, const float* __restrict__ be,
                          const float* __restrict__ m, const float* __restrict__ v,
                          const float* __restrict__ bias,
                          float* __restrict__ s, float* __restrict__ t, int n) {
  int i = blockIdx.x * 256 + threadIdx.x;
  if (i >= n) return;
  float sc = g[i] * rsqrtf(v[i] + EPS_);
  float bb = bias ? bias[i] : 0.f;
  s[i] = sc;
  t[i] = (bb - m[i]) * sc + be[i];
}

__global__ void prep_bias_k(const float* __restrict__ bias,
                            float* __restrict__ s, float* __restrict__ t, int n) {
  int i = blockIdx.x * 256 + threadIdx.x;
  if (i >= n) return;
  s[i] = 1.f;
  t[i] = bias[i];
}

// y_bf16 = bf16(x*s[c] + t[c]);   c = i % U_   (fused bno + downconvert)
__global__ void affine_bf16_k(const float* __restrict__ x, const float* __restrict__ s,
                              const float* __restrict__ t, __bf16* __restrict__ y,
                              int n) {
  int i = blockIdx.x * 256 + threadIdx.x;
  if (i >= n) return;
  int c = i & (U_ - 1);
  y[i] = (__bf16)(x[i] * s[c] + t[c]);
}

__global__ void scan_init_k(float* __restrict__ qf, __bf16* __restrict__ qb,
                            unsigned* __restrict__ sync, int n) {
  int i = blockIdx.x * 256 + threadIdx.x;
  if (i < n) { qf[i] = 0.f; qb[i] = (__bf16)0.f; }
  if (i < 8) sync[i] = 0u;
}

// ---------------------------------------------------------------------------
// Tapped implicit-GEMM conv / GEMM with fused BN + residual + PReLU epilogue.
//  Wave tile 16(M) x 64(N): one A fragment feeds 4 WMMAs.
//  Block: 256 threads = 8 waves; block tile 32(M) x 256(N).
//  Zero-padding of conv taps via zero-page pointer substitution (EXEC stays
//  all-ones; loads unconditional).
// ---------------------------------------------------------------------------
template <int CIN, int NTAPS>
__global__ __launch_bounds__(256) void conv_gemm_wmma(
    const __bf16* __restrict__ A, const __bf16* __restrict__ Bp,
    const __bf16* __restrict__ Zpage,
    const float* __restrict__ S, const float* __restrict__ Tt,
    const float* __restrict__ Pa, const float* __restrict__ Res,
    float* __restrict__ OutF, __bf16* __restrict__ OutB, int N)
{
  constexpr int KTOT = CIN * NTAPS;
  const int lane = threadIdx.x & 31;
  const int wave = threadIdx.x >> 5;
  const int lrow = lane & 15;
  const int lhi  = lane >> 4;

  const int tile_m = blockIdx.y * 32 + (wave & 1) * 16;
  const int tile_n = blockIdx.x * 256 + (wave >> 1) * 64;

  const int m_g   = tile_m + lrow;          // this lane's A row
  const int t_idx = m_g & (T_ - 1);         // time index within sequence

  // Per-tap A row pointers; invalid taps read the zero page (loop-invariant).
  const __bf16* arow[NTAPS];
#pragma unroll
  for (int dk = 0; dk < NTAPS; ++dk) {
    const int offt = dk - (NTAPS >> 1);
    const bool valid = ((unsigned)(t_idx + offt) < (unsigned)T_);
    arow[dk] = valid ? (A + (size_t)(m_g + offt) * CIN + (lhi << 4)) : Zpage;
  }

  const __bf16* brow[4];
#pragma unroll
  for (int j = 0; j < 4; ++j)
    brow[j] = Bp + (size_t)(tile_n + j * 16 + lrow) * KTOT + (lhi << 4);

  v8f acc[4] = {{}, {}, {}, {}};
#pragma unroll
  for (int dk = 0; dk < NTAPS; ++dk) {
#pragma unroll 4
    for (int k = 0; k < CIN; k += 32) {
      const v16bf af = load_a_frag(arow[dk] + k);
      const int kb = dk * CIN + k;
#pragma unroll
      for (int j = 0; j < 4; ++j) {
        const v16bf bf = *(const v16bf*)(brow[j] + kb);
        acc[j] = wmma_bf16(af, bf, acc[j]);
      }
    }
  }

  // Epilogue: acc[j][v] is C[tile_m + v + 8*lhi][tile_n + j*16 + lrow]
  const int row0 = tile_m + (lhi << 3);
#pragma unroll
  for (int j = 0; j < 4; ++j) {
    const int col = tile_n + j * 16 + lrow;
    const float sc = S[col];
    const float tc = Tt[col];
    const float pa = Pa ? Pa[col] : 0.f;
    const size_t base = (size_t)row0 * N + col;
#pragma unroll
    for (int v = 0; v < 8; ++v) {
      float y = acc[j][v] * sc + tc;
      if (Res) y += Res[base + (size_t)v * N];
      if (Pa)  y = fmaxf(y, 0.f) + pa * fminf(y, 0.f);
      if (OutF) OutF[base + (size_t)v * N] = y;
      if (OutB) OutB[base + (size_t)v * N] = (__bf16)y;
    }
  }
}

// ---------------------------------------------------------------------------
// Device-wide spin barrier for the persistent scan kernel
// ---------------------------------------------------------------------------
__device__ __forceinline__ void grid_barrier(unsigned* cnt, unsigned* gen, unsigned nb) {
  __syncthreads();
  __threadfence();
  if (threadIdx.x == 0) {
    unsigned g = atomicAdd(gen, 0u);
    if (atomicAdd(cnt, 1u) == nb - 1u) {
      atomicExch(cnt, 0u);
      __threadfence();
      atomicAdd(gen, 1u);
    } else {
      while (atomicAdd(gen, 0u) == g) { __builtin_amdgcn_s_sleep(2); }
    }
  }
  __syncthreads();
}

// ---------------------------------------------------------------------------
// Residual GRU scan (persistent, NB_SCAN blocks x 8 waves = 128 waves).
// Recurrent weights staged once into LDS (96 KB/block of 320 KB/WGP); per-step
// WMMA B operands come from ds_load. z/r epilogue split is wave-uniform
// (each 16-wide N tile lies entirely in the z or r half).
// ---------------------------------------------------------------------------
#define SCAN_LDS_BYTES ((64 + 32) * U_ * 2)   // 98304 bytes

__global__ __launch_bounds__(256) void gru_scan_wmma(
    const float* __restrict__ Xzrq,
    const __bf16* __restrict__ Uzr,   // [1024][512] packed
    const __bf16* __restrict__ Uq,    // [512][512] packed
    float* __restrict__ qf, __bf16* __restrict__ qb,
    __bf16* __restrict__ rqb, float* __restrict__ zbuf,
    float* __restrict__ outp, int return_seq,
    unsigned* __restrict__ sync)
{
  extern __shared__ __bf16 smem[];
  __bf16* uzr_s = smem;               // 64 x 512
  __bf16* uq_s  = smem + 64 * U_;     // 32 x 512

  const int lane  = threadIdx.x & 31;
  const int wave  = threadIdx.x >> 5;
  const int waveg = blockIdx.x * 8 + wave;   // 0..127
  const int lrow  = lane & 15;
  const int lhi   = lane >> 4;
  unsigned* cnt = sync;
  unsigned* gen = sync + 1;

  // ---- one-time stage of this block's recurrent-weight slice into LDS ----
  {
    const uint4* g1 = (const uint4*)(Uzr + (size_t)(blockIdx.x * 64) * U_);
    uint4* l1 = (uint4*)uzr_s;
    for (int i = threadIdx.x; i < 64 * U_ / 8; i += 256) l1[i] = g1[i];
    const uint4* g2 = (const uint4*)(Uq + (size_t)(blockIdx.x * 32) * U_);
    uint4* l2 = (uint4*)uq_s;
    for (int i = threadIdx.x; i < 32 * U_ / 8; i += 256) l2[i] = g2[i];
  }
  __syncthreads();

  const int tm = (waveg & 3) * 16;           // batch-row tile (0..48)
  const int b0 = tm + (lhi << 3);            // first batch row this lane touches
  // LDS row bases (block-local)
  const __bf16* bA0 = uzr_s + (size_t)((wave >> 2) * 32 + lrow)      * U_ + (lhi << 4);
  const __bf16* bA1 = uzr_s + (size_t)((wave >> 2) * 32 + 16 + lrow) * U_ + (lhi << 4);
  const __bf16* bB  = uq_s  + (size_t)((wave >> 2) * 16 + lrow)      * U_ + (lhi << 4);

  constexpr size_t XROW = (size_t)T_ * 1536;   // Xzrq stride per batch row

  for (int t = 0; t < T_; ++t) {
    // ---------------- Phase A: z and r gates ----------------
    {
      const int tn0 = (waveg >> 2) * 32;     // two N tiles: tn0, tn0+16
      const __bf16* aptr = qb + (size_t)(tm + lrow) * U_ + (lhi << 4);
      v8f acc0 = {}, acc1 = {};
#pragma unroll 4
      for (int k = 0; k < U_; k += 32) {
        const v16bf af = load_a_frag(aptr + k);
        acc0 = wmma_bf16(af, *(const v16bf*)(bA0 + k), acc0);
        acc1 = wmma_bf16(af, *(const v16bf*)(bA1 + k), acc1);
      }
#pragma unroll
      for (int tile = 0; tile < 2; ++tile) {
        const v8f acc = tile ? acc1 : acc0;
        const int tn_tile = tn0 + tile * 16;           // wave-uniform
        const int n = tn_tile + lrow;
        const float* xb = Xzrq + (size_t)b0 * XROW + (size_t)t * 1536 + n;
        if (tn_tile < U_) {                            // z half (uniform branch)
          float* zp = zbuf + b0 * U_ + n;
#pragma unroll
          for (int v = 0; v < 8; ++v)
            zp[v * U_] = fast_sigmoid(acc[v] + xb[(size_t)v * XROW]);
        } else {                                       // r half (uniform branch)
          const int nn = n - U_;
          const float* qp = qf + b0 * U_ + nn;
          __bf16* rp = rqb + b0 * U_ + nn;
#pragma unroll
          for (int v = 0; v < 8; ++v) {
            const float r = fast_sigmoid(acc[v] + xb[(size_t)v * XROW]);
            rp[v * U_] = (__bf16)(r * qp[v * U_]);
          }
        }
      }
    }
    grid_barrier(cnt, gen, NB_SCAN);

    // ---------------- Phase B: candidate + state update ----------------
    {
      const int tn = (waveg >> 2) * 16;      // 0..496
      const __bf16* aptr = rqb + (size_t)(tm + lrow) * U_ + (lhi << 4);
      v8f acc = {};
#pragma unroll 4
      for (int k = 0; k < U_; k += 32) {
        acc = wmma_bf16(load_a_frag(aptr + k), *(const v16bf*)(bB + k), acc);
      }
      const int n = tn + lrow;
      const float* xq = Xzrq + (size_t)b0 * XROW + (size_t)t * 1536 + 1024 + n;
      float* zp  = zbuf + b0 * U_ + n;
      float* qp  = qf   + b0 * U_ + n;
      __bf16* qbp = qb  + b0 * U_ + n;
      float* op  = outp + (size_t)b0 * T_ * U_ + (size_t)t * U_ + n;   // seq mode
#pragma unroll
      for (int v = 0; v < 8; ++v) {
        const float qt = fast_tanh(acc[v] + xq[(size_t)v * XROW]);
        const float z  = zp[v * U_];
        const float qo = qp[v * U_];
        const float qn = (1.f - z) * qo + z * qt + ALPHA_ * z * fast_tanh(1.f - z);
        qp[v * U_]  = qn;
        qbp[v * U_] = (__bf16)qn;
        if (return_seq)        op[(size_t)v * T_ * U_] = qn;
        else if (t == T_ - 1)  outp[(b0 + v) * U_ + n] = qn;
      }
    }
    grid_barrier(cnt, gen, NB_SCAN);
  }
}

// ---------------------------------------------------------------------------
// Host driver
// ---------------------------------------------------------------------------
extern "C" void kernel_launch(void* const* d_in, const int* in_sizes, int n_in,
                              void* d_out, int out_size, void* d_ws, size_t ws_size,
                              hipStream_t stream)
{
  (void)in_sizes; (void)n_in; (void)out_size; (void)ws_size;
  char* wsb = (char*)d_ws;
  size_t off = 0;
  auto allocB = [&](size_t nbytes) -> void* {
    off = (off + 255) & ~(size_t)255;
    void* p = wsb + off;
    off += nbytes;
    return p;
  };
  auto allocBF = [&](size_t nel) { return (__bf16*)allocB(nel * 2); };
  auto allocF  = [&](size_t nel) { return (float*)allocB(nel * 4); };

  // Input index map (setup_inputs flatten order)
  const int L0 = 1, L1 = 37;
  auto in = [&](int i) { return (const float*)d_in[i]; };

  // ---- packed bf16 weights (B^T layout [N][K]) ----
  __bf16* bp_c1_0 = allocBF(768ull * 512);
  __bf16* bp_c2_0 = allocBF(1536ull * 512);
  __bf16* bp_c3_0 = allocBF(512ull * 512);
  __bf16* bp_pj_0 = allocBF(256ull * 512);
  __bf16* bp_w_0  = allocBF(512ull * 1536);   // Wz|Wr|Wq ^T
  __bf16* bp_uzr_0= allocBF(512ull * 1024);   // Uz|Ur ^T
  __bf16* bp_uq_0 = allocBF(512ull * 512);
  __bf16* bp_c1_1 = allocBF(1536ull * 512);
  __bf16* bp_c2_1 = allocBF(1536ull * 512);
  __bf16* bp_c3_1 = allocBF(512ull * 512);
  __bf16* bp_pj_1 = allocBF(512ull * 512);
  __bf16* bp_w_1  = allocBF(512ull * 1536);
  __bf16* bp_uzr_1= allocBF(512ull * 1024);
  __bf16* bp_uq_1 = allocBF(512ull * 512);
  __bf16* zpage   = allocBF(1024);            // zero page for conv padding

  // ---- epilogue (s,t) coefficient arrays ----
  float* s_pj0 = allocF(512);  float* t_pj0 = allocF(512);
  float* s_b1_0= allocF(512);  float* t_b1_0= allocF(512);
  float* s_b2_0= allocF(512);  float* t_b2_0= allocF(512);
  float* s_b3_0= allocF(512);  float* t_b3_0= allocF(512);
  float* s_zr0 = allocF(1536); float* t_zr0 = allocF(1536);
  float* s_bno = allocF(512);  float* t_bno = allocF(512);
  float* s_pj1 = allocF(512);  float* t_pj1 = allocF(512);
  float* s_b1_1= allocF(512);  float* t_b1_1= allocF(512);
  float* s_b2_1= allocF(512);  float* t_b2_1= allocF(512);
  float* s_b3_1= allocF(512);  float* t_b3_1= allocF(512);
  float* s_zr1 = allocF(1536); float* t_zr1 = allocF(1536);

  // ---- activation buffers ----
  __bf16* abuf0 = allocBF((size_t)M_ * U_);   // layer input (bf16)
  __bf16* abuf1 = allocBF((size_t)M_ * U_);   // o1 / y
  __bf16* abuf2 = allocBF((size_t)M_ * U_);   // o2
  float*  f32buf= allocF((size_t)M_ * U_);    // shortcut, later GRU seq out
  float*  xzrq  = allocF((size_t)M_ * 1536);  // GRU input projections
  float*  qf    = allocF(B_ * U_);
  __bf16* qb    = allocBF(B_ * U_);
  __bf16* rqb   = allocBF(B_ * U_);
  float*  zb    = allocF(B_ * U_);
  unsigned* syncp = (unsigned*)allocB(256);

  auto nblk = [](size_t n) { return (unsigned)((n + 255) / 256); };
  auto pack = [&](const float* w, __bf16* dst, int K, int N) {
    pack_transpose_k<<<nblk((size_t)K * N), 256, 0, stream>>>(w, dst, K, N);
  };
  auto gemm = [&](const __bf16* A, const __bf16* Bp, const float* S, const float* Tt,
                  const float* Pa, const float* Res, float* OF, __bf16* OB,
                  int N, int Cin, int ntaps) {
    dim3 g(N / 256, M_ / 32);
    if (Cin == 256 && ntaps == 1)
      conv_gemm_wmma<256, 1><<<g, 256, 0, stream>>>(A, Bp, zpage, S, Tt, Pa, Res, OF, OB, N);
    else if (Cin == 256 && ntaps == 3)
      conv_gemm_wmma<256, 3><<<g, 256, 0, stream>>>(A, Bp, zpage, S, Tt, Pa, Res, OF, OB, N);
    else if (Cin == 512 && ntaps == 1)
      conv_gemm_wmma<512, 1><<<g, 256, 0, stream>>>(A, Bp, zpage, S, Tt, Pa, Res, OF, OB, N);
    else
      conv_gemm_wmma<512, 3><<<g, 256, 0, stream>>>(A, Bp, zpage, S, Tt, Pa, Res, OF, OB, N);
  };

  // ================= weight packing (both layers) =================
  zero_bf16_k<<<4, 256, 0, stream>>>(zpage, 1024);
  pack(in(L0 + 0), bp_c1_0, 3 * 256, 512);
  pack(in(L0 + 2), bp_c2_0, 3 * 512, 512);
  pack(in(L0 + 4), bp_c3_0, 512, 512);
  pack(in(L0 + 6), bp_pj_0, 256, 512);
  pack(in(L0 + 27), bp_w_0,               512, 512);   // Wz
  pack(in(L0 + 30), bp_w_0 + 262144,      512, 512);   // Wr
  pack(in(L0 + 33), bp_w_0 + 524288,      512, 512);   // Wq
  pack(in(L0 + 28), bp_uzr_0,             512, 512);   // Uz
  pack(in(L0 + 31), bp_uzr_0 + 262144,    512, 512);   // Ur
  pack(in(L0 + 34), bp_uq_0,              512, 512);   // Uq

  pack(in(L1 + 0), bp_c1_1, 3 * 512, 512);
  pack(in(L1 + 2), bp_c2_1, 3 * 512, 512);
  pack(in(L1 + 4), bp_c3_1, 512, 512);
  pack(in(L1 + 6), bp_pj_1, 512, 512);
  pack(in(L1 + 23), bp_w_1,               512, 512);
  pack(in(L1 + 26), bp_w_1 + 262144,      512, 512);
  pack(in(L1 + 29), bp_w_1 + 524288,      512, 512);
  pack(in(L1 + 24), bp_uzr_1,             512, 512);
  pack(in(L1 + 27), bp_uzr_1 + 262144,    512, 512);
  pack(in(L1 + 30), bp_uq_1,              512, 512);

  // ================= epilogue coefficients =================
  prep_bias_k<<<2, 256, 0, stream>>>(in(L0 + 7), s_pj0, t_pj0, 512);
  prep_bn_k<<<2, 256, 0, stream>>>(in(L0+8),  in(L0+9),  in(L0+10), in(L0+11), in(L0+1), s_b1_0, t_b1_0, 512);
  prep_bn_k<<<2, 256, 0, stream>>>(in(L0+12), in(L0+13), in(L0+14), in(L0+15), in(L0+3), s_b2_0, t_b2_0, 512);
  prep_bn_k<<<2, 256, 0, stream>>>(in(L0+16), in(L0+17), in(L0+18), in(L0+19), in(L0+5), s_b3_0, t_b3_0, 512);
  prep_bn_k<<<2, 256, 0, stream>>>(in(L0+20), in(L0+21), in(L0+22), in(L0+23), nullptr,  s_bno,  t_bno,  512);
  prep_bias_k<<<2, 256, 0, stream>>>(in(L0 + 29), s_zr0,        t_zr0,        512);  // bz
  prep_bias_k<<<2, 256, 0, stream>>>(in(L0 + 32), s_zr0 + 512,  t_zr0 + 512,  512);  // br
  prep_bias_k<<<2, 256, 0, stream>>>(in(L0 + 35), s_zr0 + 1024, t_zr0 + 1024, 512);  // bq

  prep_bias_k<<<2, 256, 0, stream>>>(in(L1 + 7), s_pj1, t_pj1, 512);
  prep_bn_k<<<2, 256, 0, stream>>>(in(L1+8),  in(L1+9),  in(L1+10), in(L1+11), in(L1+1), s_b1_1, t_b1_1, 512);
  prep_bn_k<<<2, 256, 0, stream>>>(in(L1+12), in(L1+13), in(L1+14), in(L1+15), in(L1+3), s_b2_1, t_b2_1, 512);
  prep_bn_k<<<2, 256, 0, stream>>>(in(L1+16), in(L1+17), in(L1+18), in(L1+19), in(L1+5), s_b3_1, t_b3_1, 512);
  prep_bias_k<<<2, 256, 0, stream>>>(in(L1 + 25), s_zr1,        t_zr1,        512);
  prep_bias_k<<<2, 256, 0, stream>>>(in(L1 + 28), s_zr1 + 512,  t_zr1 + 512,  512);
  prep_bias_k<<<2, 256, 0, stream>>>(in(L1 + 31), s_zr1 + 1024, t_zr1 + 1024, 512);

  // ================= Layer 0 =================
  f32_to_bf16_k<<<nblk((size_t)M_ * F_), 256, 0, stream>>>((const float*)d_in[0], abuf0,
                                                           M_ * F_);
  // shortcut = x @ proj + b                -> f32buf
  gemm(abuf0, bp_pj_0, s_pj0, t_pj0, nullptr, nullptr, f32buf, nullptr, 512, 256, 1);
  // o1 = PReLU(BN(conv1(x)))              -> abuf1 (bf16)
  gemm(abuf0, bp_c1_0, s_b1_0, t_b1_0, in(L0 + 24), nullptr, nullptr, abuf1, 512, 256, 3);
  // o2 = PReLU(BN(conv2(o1)))             -> abuf2
  gemm(abuf1, bp_c2_0, s_b2_0, t_b2_0, in(L0 + 25), nullptr, nullptr, abuf2, 512, 512, 3);
  // y = PReLU(BN(conv3(o2)) + shortcut)   -> abuf1
  gemm(abuf2, bp_c3_0, s_b3_0, t_b3_0, in(L0 + 26), f32buf, nullptr, abuf1, 512, 512, 1);
  // Xz|Xr|Xq = y @ (Wz|Wr|Wq) + b         -> xzrq (f32)
  gemm(abuf1, bp_w_0, s_zr0, t_zr0, nullptr, nullptr, xzrq, nullptr, 1536, 512, 1);
  // GRU scan (return_seq) -> f32buf
  scan_init_k<<<nblk(B_ * U_), 256, 0, stream>>>(qf, qb, syncp, B_ * U_);
  gru_scan_wmma<<<NB_SCAN, 256, SCAN_LDS_BYTES, stream>>>(xzrq, bp_uzr_0, bp_uq_0, qf, qb,
                                                          rqb, zb, f32buf, 1, syncp);
  // bno + bf16 downconvert -> abuf0 (layer1 input)
  affine_bf16_k<<<nblk((size_t)M_ * U_), 256, 0, stream>>>(f32buf, s_bno, t_bno, abuf0,
                                                           M_ * U_);

  // ================= Layer 1 =================
  gemm(abuf0, bp_pj_1, s_pj1, t_pj1, nullptr, nullptr, f32buf, nullptr, 512, 512, 1);
  gemm(abuf0, bp_c1_1, s_b1_1, t_b1_1, in(L1 + 20), nullptr, nullptr, abuf1, 512, 512, 3);
  gemm(abuf1, bp_c2_1, s_b2_1, t_b2_1, in(L1 + 21), nullptr, nullptr, abuf2, 512, 512, 3);
  gemm(abuf2, bp_c3_1, s_b3_1, t_b3_1, in(L1 + 22), f32buf, nullptr, abuf1, 512, 512, 1);
  gemm(abuf1, bp_w_1, s_zr1, t_zr1, nullptr, nullptr, xzrq, nullptr, 1536, 512, 1);
  scan_init_k<<<nblk(B_ * U_), 256, 0, stream>>>(qf, qb, syncp, B_ * U_);
  // final state only -> d_out [64,512] f32
  gru_scan_wmma<<<NB_SCAN, 256, SCAN_LDS_BYTES, stream>>>(xzrq, bp_uzr_1, bp_uq_1, qf, qb,
                                                          rqb, zb, (float*)d_out, 0, syncp);
}
